// SimpleSetAbstraction_55456617726261
// MI455X (gfx1250) — matrix operand order repaired and checked by reading, not compiled, not run on hardware
//
#include <hip/hip_runtime.h>

#define BATCH   8
#define NPTS    16384
#define DPTS    64
#define NPOINT  1024
#define NSAMPLE 32
#define RAD2    0.04f
#define CIN0    67
#define KPAD0   96
#define C3      128
#define MROWS   (BATCH * NPOINT * NSAMPLE)   // 262144
#define EPSBN   1e-5f

typedef __attribute__((ext_vector_type(16))) __bf16 v16bf;
typedef __attribute__((ext_vector_type(8)))  __bf16 v8bf;
typedef __attribute__((ext_vector_type(8)))  float  v8f;

__device__ __forceinline__ unsigned short f2bf_bits(float f) {
    unsigned u = __builtin_bit_cast(unsigned, f);
    u += 0x7FFFu + ((u >> 16) & 1u);          // round-to-nearest-even
    return (unsigned short)(u >> 16);
}

// ---------------------------------------------------------------------------
// Phase 1: farthest point sampling. One block per batch; point coords and the
// running min-distance live entirely in registers (16 pts / thread).
// ---------------------------------------------------------------------------
__global__ __launch_bounds__(1024) void fps_kernel(
    const float* __restrict__ xyz, int* __restrict__ fps_idx,
    float* __restrict__ nxyz, float* __restrict__ out0) {
    const int b = blockIdx.x;
    const int t = threadIdx.x;
    const float* px = xyz + (size_t)b * 3 * NPTS;
    float lx[16], ly[16], lz[16], ld[16];
#pragma unroll
    for (int c = 0; c < 16; ++c) {
        int j = t + 1024 * c;
        lx[c] = px[j]; ly[c] = px[NPTS + j]; lz[c] = px[2 * NPTS + j];
        ld[c] = 1e10f;
    }
    __shared__ float redv[32];
    __shared__ int   redi[32];
    __shared__ int   scur;
    if (t == 0) scur = 0;
    __syncthreads();
    for (int it = 0; it < NPOINT; ++it) {
        int curr = scur;
        float cx = px[curr], cy = px[NPTS + curr], cz = px[2 * NPTS + curr];
        if (t == 0) {
            fps_idx[b * NPOINT + it] = curr;
            float* np_ = nxyz + ((size_t)b * NPOINT + it) * 3;
            np_[0] = cx; np_[1] = cy; np_[2] = cz;
            float* o = out0 + (size_t)b * 3 * NPOINT + it;
            o[0] = cx; o[NPOINT] = cy; o[2 * NPOINT] = cz;
        }
        float bv = -1.0f; int bi = 0;
#pragma unroll
        for (int c = 0; c < 16; ++c) {
            int j = t + 1024 * c;
            float dx = lx[c] - cx, dy = ly[c] - cy, dz = lz[c] - cz;
            float d  = dx * dx + dy * dy + dz * dz;
            float dm = fminf(ld[c], d);
            ld[c] = dm;
            if (dm > bv) { bv = dm; bi = j; }
        }
#pragma unroll
        for (int off = 16; off > 0; off >>= 1) {
            float ov = __shfl_down(bv, off);
            int   oi = __shfl_down(bi, off);
            if (ov > bv || (ov == bv && oi < bi)) { bv = ov; bi = oi; }
        }
        if ((t & 31) == 0) { redv[t >> 5] = bv; redi[t >> 5] = bi; }
        __syncthreads();
        if (t < 32) {
            bv = redv[t]; bi = redi[t];
#pragma unroll
            for (int off = 16; off > 0; off >>= 1) {
                float ov = __shfl_down(bv, off);
                int   oi = __shfl_down(bi, off);
                if (ov > bv || (ov == bv && oi < bi)) { bv = ov; bi = oi; }
            }
            if (t == 0) scur = bi;
        }
        __syncthreads();
    }
}

// ---------------------------------------------------------------------------
// Phase 2: ball query. One wave32 per query point; ballot-based stream
// compaction to take the first <=32 ascending indices with d^2 <= r^2.
// ---------------------------------------------------------------------------
__global__ __launch_bounds__(256) void ballquery_kernel(
    const float* __restrict__ xyz, const float* __restrict__ nxyz,
    int* __restrict__ gidx) {
    int gw   = (int)((blockIdx.x * blockDim.x + threadIdx.x) >> 5);
    int lane = threadIdx.x & 31;
    int b = gw >> 10, s = gw & 1023;
    const float* px = xyz + (size_t)b * 3 * NPTS;
    const float* ctr = nxyz + ((size_t)b * NPOINT + s) * 3;
    float cx = ctr[0], cy = ctr[1], cz = ctr[2];
    int* out = gidx + ((size_t)b * NPOINT + s) * NSAMPLE;
    int count = 0, first = 0;
    for (int c0 = 0; c0 < NPTS; c0 += 32) {
        int j = c0 + lane;
        float dx = px[j] - cx, dy = px[NPTS + j] - cy, dz = px[2 * NPTS + j] - cz;
        int pred = (dx * dx + dy * dy + dz * dz) <= RAD2;
        unsigned mask = (unsigned)__ballot(pred);
        if (count == 0 && mask) first = c0 + __ffs((int)mask) - 1;
        if (pred) {
            int pos = count + __popc(mask & ((1u << lane) - 1u));
            if (pos < NSAMPLE) out[pos] = j;
        }
        count += __popc(mask);
        if (count >= NSAMPLE) break;
    }
    if (count < NSAMPLE && lane < NSAMPLE - count) out[count + lane] = first;
}

// ---------------------------------------------------------------------------
// Phase 3: gather + concat -> bf16 F0 (MROWS x KPAD0), zero-padded K.
// ---------------------------------------------------------------------------
__global__ void gather_kernel(
    const float* __restrict__ xyz, const float* __restrict__ pts,
    const float* __restrict__ nxyz, const int* __restrict__ gidx,
    unsigned short* __restrict__ F0) {
    int row = blockIdx.x;          // (b*1024+s)*32+k
    int t   = threadIdx.x;         // 0..95
    int bs  = row >> 5;
    int b   = bs >> 10;
    int gi  = gidx[(size_t)row];
    float val = 0.0f;
    if (t < 3)
        val = xyz[(size_t)b * 3 * NPTS + (size_t)t * NPTS + gi] - nxyz[(size_t)bs * 3 + t];
    else if (t < CIN0)
        val = pts[((size_t)b * DPTS + (t - 3)) * NPTS + gi];
    F0[(size_t)row * KPAD0 + t] = f2bf_bits(val);
}

// ---------------------------------------------------------------------------
// Weight (Cout x Cin, f32) -> bf16 in B *fragment* order so each lane's 16
// elements are a contiguous 32-byte run:
//   Wbf[((kc*NT + nt)*32 + lane)*16 + e] = W[nt*16 + lane%16][kc*32 + (lane/16)*16 + e]
// (zero-padded for k >= cin). grid.x = KC*NT, block = 512 (lane*16 + e).
// ---------------------------------------------------------------------------
__global__ __launch_bounds__(512) void wfrag_kernel(
    const float* __restrict__ W, unsigned short* __restrict__ Wbf,
    int cin, int nt_count) {
    int blk  = blockIdx.x;            // kc*NT + nt
    int kc   = blk / nt_count;
    int nt   = blk - kc * nt_count;
    int t    = threadIdx.x;           // lane*16 + e
    int lane = t >> 4, e = t & 15;
    int k = kc * 32 + (lane >> 4) * 16 + e;
    int o = nt * 16 + (lane & 15);
    float v = (k < cin) ? W[(size_t)o * cin + k] : 0.0f;
    Wbf[(size_t)blk * 512 + t] = f2bf_bits(v);
}

// ---------------------------------------------------------------------------
// Phase 4: bf16 WMMA GEMM, Y = X(bf16) @ W + bias, plus per-channel
// sum / sum-of-squares for global batch-norm. Each wave: 16 rows x NT*16 cols.
// A fragments: two contiguous b128 loads per k-chunk. B fragments: one
// contiguous 32B load from the pre-swizzled weight buffer.
// ---------------------------------------------------------------------------
template <int NT, int KC>
__global__ __launch_bounds__(256) void gemm_kernel(
    const unsigned short* __restrict__ X, const unsigned short* __restrict__ Wbf,
    const float* __restrict__ bias, float* __restrict__ Y,
    float* __restrict__ gsum, float* __restrict__ gsumsq) {
    constexpr int NOUT = NT * 16;
    constexpr int KPAD = KC * 32;
    __shared__ float lsum[NOUT], lsq[NOUT];
    int t = threadIdx.x;
    if (t < NOUT) { lsum[t] = 0.f; lsq[t] = 0.f; }
    __syncthreads();
    int wave = t >> 5, lane = t & 31;
    int half = lane >> 4, l16 = lane & 15;
    size_t mbase = ((size_t)blockIdx.x * 8 + wave) * 16;
    v8f acc[NT];
#pragma unroll
    for (int nt = 0; nt < NT; ++nt)
        acc[nt] = (v8f){0.f, 0.f, 0.f, 0.f, 0.f, 0.f, 0.f, 0.f};
    const unsigned short* Arow = X + (mbase + (size_t)l16) * KPAD;
    const v16bf* Bfrag = (const v16bf*)(Wbf + (size_t)lane * 16);
#pragma unroll
    for (int kc = 0; kc < KC; ++kc) {
        int kbase = kc * 32;
        // A fragment 16x32 bf16: lane%16 = row M; this lane-half needs K run
        // [kbase+16h, +16) -> two aligned b128 loads.
        v8bf alo = *(const v8bf*)(Arow + kbase + half * 16);
        v8bf ahi = *(const v8bf*)(Arow + kbase + half * 16 + 8);
        v16bf a = __builtin_shufflevector(alo, ahi, 0, 1, 2, 3, 4, 5, 6, 7,
                                          8, 9, 10, 11, 12, 13, 14, 15);
#pragma unroll
        for (int nt = 0; nt < NT; ++nt) {
            v16bf bm = Bfrag[(size_t)(kc * NT + nt) * 32];
            acc[nt] = __builtin_amdgcn_wmma_f32_16x16x32_bf16(
                false, a, false, bm, (short)0, acc[nt], false, false);
        }
    }
    // C layout: VGPR r -> M = r + 8*half, N = lane%16.
#pragma unroll
    for (int nt = 0; nt < NT; ++nt) {
        int n = nt * 16 + l16;
        float bsv = bias[n];
        float s = 0.f, s2 = 0.f;
#pragma unroll
        for (int r = 0; r < 8; ++r) {
            float y = acc[nt][r] + bsv;
            Y[(mbase + (size_t)(r + 8 * half)) * NOUT + n] = y;
            s += y; s2 += y * y;
        }
        atomicAdd(&lsum[n], s);
        atomicAdd(&lsq[n], s2);
    }
    __syncthreads();
    if (t < NOUT) {
        atomicAdd(&gsum[t], lsum[t]);
        atomicAdd(&gsumsq[t], lsq[t]);
    }
}

// BN finalize: scale = g/sqrt(var+eps), shift = beta - mean*scale.
__global__ void finalize_kernel(const float* __restrict__ sum,
                                const float* __restrict__ sumsq,
                                const float* __restrict__ g,
                                const float* __restrict__ be,
                                float* __restrict__ scale,
                                float* __restrict__ shift, int nout) {
    int n = threadIdx.x;
    if (n < nout) {
        float inv  = 1.0f / (float)MROWS;
        float mean = sum[n] * inv;
        float var  = sumsq[n] * inv - mean * mean;
        float a    = g[n] * rsqrtf(var + EPSBN);
        scale[n] = a;
        shift[n] = be[n] - mean * a;
    }
}

// Elementwise BN + ReLU + recode to bf16 for the next GEMM.
__global__ void bnrelu_kernel(const float* __restrict__ Y,
                              unsigned short* __restrict__ X,
                              const float* __restrict__ scale,
                              const float* __restrict__ shift,
                              int nmask, size_t total) {
    size_t i = (size_t)blockIdx.x * blockDim.x + threadIdx.x;
    if (i < total) {
        int n = (int)(i & (size_t)nmask);
        X[i] = f2bf_bits(fmaxf(Y[i] * scale[n] + shift[n], 0.0f));
    }
}

// Final layer: BN + ReLU + max over k + transposed (B,128,NPOINT) store.
__global__ __launch_bounds__(C3) void maxpool_kernel(
    const float* __restrict__ Y, const float* __restrict__ scale,
    const float* __restrict__ shift, float* __restrict__ out2) {
    int bs = blockIdx.x;          // b*1024+s
    int o  = threadIdx.x;         // 0..127
    int b = bs >> 10, s = bs & 1023;
    float a = scale[o], c = shift[o];
    float m = 0.0f;               // ReLU output is >= 0
    size_t base = (size_t)bs * NSAMPLE * C3 + o;
#pragma unroll 4
    for (int k = 0; k < NSAMPLE; ++k) {
        float y = Y[base + (size_t)k * C3] * a + c;
        m = fmaxf(m, y);
    }
    out2[(size_t)b * C3 * NPOINT + (size_t)o * NPOINT + s] = m;
}

// ---------------------------------------------------------------------------
extern "C" void kernel_launch(void* const* d_in, const int* in_sizes, int n_in,
                              void* d_out, int out_size, void* d_ws, size_t ws_size,
                              hipStream_t stream) {
    (void)in_sizes; (void)n_in; (void)out_size; (void)ws_size;
    const float* xyz = (const float*)d_in[0];
    const float* pts = (const float*)d_in[1];
    const float* W1  = (const float*)d_in[2];
    const float* W2  = (const float*)d_in[3];
    const float* W3  = (const float*)d_in[4];
    const float* cb1 = (const float*)d_in[5];
    const float* cb2 = (const float*)d_in[6];
    const float* cb3 = (const float*)d_in[7];
    const float* g1  = (const float*)d_in[8];
    const float* g2  = (const float*)d_in[9];
    const float* g3  = (const float*)d_in[10];
    const float* be1 = (const float*)d_in[11];
    const float* be2 = (const float*)d_in[12];
    const float* be3 = (const float*)d_in[13];

    char* ws = (char*)d_ws;
    size_t off = 0;
    auto alloc = [&](size_t bytes) -> void* {
        void* p = ws + off;
        off = (off + bytes + 255) & ~(size_t)255;
        return p;
    };
    int*            fps_idx = (int*)alloc((size_t)BATCH * NPOINT * 4);
    float*          nxyz    = (float*)alloc((size_t)BATCH * NPOINT * 3 * 4);
    int*            gidx    = (int*)alloc((size_t)MROWS * 4);
    float*          stats   = (float*)alloc(3 * 512 * sizeof(float));
    unsigned short* Wb1     = (unsigned short*)alloc((size_t)3 * 4 * 512 * 2);  // KC=3,NT=4
    unsigned short* Wb2     = (unsigned short*)alloc((size_t)2 * 4 * 512 * 2);  // KC=2,NT=4
    unsigned short* Wb3     = (unsigned short*)alloc((size_t)2 * 8 * 512 * 2);  // KC=2,NT=8
    unsigned short* F0      = (unsigned short*)alloc((size_t)MROWS * KPAD0 * 2);
    unsigned short* Xb      = (unsigned short*)alloc((size_t)MROWS * 64 * 2);
    float*          Yb      = (float*)alloc((size_t)MROWS * C3 * 4);

    float* out0 = (float*)d_out;
    float* out2 = out0 + (size_t)BATCH * 3 * NPOINT;

    float* st1 = stats;
    float* st2 = stats + 512;
    float* st3 = stats + 1024;

    hipMemsetAsync(stats, 0, 3 * 512 * sizeof(float), stream);
    wfrag_kernel<<<3 * 4, 512, 0, stream>>>(W1, Wb1, CIN0, 4);
    wfrag_kernel<<<2 * 4, 512, 0, stream>>>(W2, Wb2, 64, 4);
    wfrag_kernel<<<2 * 8, 512, 0, stream>>>(W3, Wb3, 64, 8);

    fps_kernel<<<BATCH, 1024, 0, stream>>>(xyz, fps_idx, nxyz, out0);
    ballquery_kernel<<<(BATCH * NPOINT * 32) / 256, 256, 0, stream>>>(xyz, nxyz, gidx);
    gather_kernel<<<MROWS, 96, 0, stream>>>(xyz, pts, nxyz, gidx, F0);

    // Layer 1: 96 -> 64
    gemm_kernel<4, 3><<<MROWS / 128, 256, 0, stream>>>(F0, Wb1, cb1, Yb, st1, st1 + 128);
    finalize_kernel<<<1, 128, 0, stream>>>(st1, st1 + 128, g1, be1, st1 + 256, st1 + 384, 64);
    bnrelu_kernel<<<(unsigned)((size_t)MROWS * 64 / 256), 256, 0, stream>>>(
        Yb, Xb, st1 + 256, st1 + 384, 63, (size_t)MROWS * 64);

    // Layer 2: 64 -> 64
    gemm_kernel<4, 2><<<MROWS / 128, 256, 0, stream>>>(Xb, Wb2, cb2, Yb, st2, st2 + 128);
    finalize_kernel<<<1, 128, 0, stream>>>(st2, st2 + 128, g2, be2, st2 + 256, st2 + 384, 64);
    bnrelu_kernel<<<(unsigned)((size_t)MROWS * 64 / 256), 256, 0, stream>>>(
        Yb, Xb, st2 + 256, st2 + 384, 63, (size_t)MROWS * 64);

    // Layer 3: 64 -> 128, fused BN+ReLU+maxpool+transpose
    gemm_kernel<8, 2><<<MROWS / 128, 256, 0, stream>>>(Xb, Wb3, cb3, Yb, st3, st3 + 128);
    finalize_kernel<<<1, 128, 0, stream>>>(st3, st3 + 128, g3, be3, st3 + 256, st3 + 384, C3);
    maxpool_kernel<<<BATCH * NPOINT, C3, 0, stream>>>(Yb, st3 + 256, st3 + 384, out2);
}
